// ContextualAttention2D_19318762897558
// MI455X (gfx1250) — compile-verified, matrix operand order checked by reading, not run on hardware
//
#include <hip/hip_runtime.h>

// ---------------------------------------------------------------------------
// ContextualAttention2D for MI455X (gfx1250, wave32, WMMA f32_16x16x32_f16)
// B=16, C=512, H*W=N=1024, L=512, CTX_DIM=768, NH=8, HD=64
// ---------------------------------------------------------------------------

typedef _Float16 h16;
typedef __attribute__((ext_vector_type(16))) _Float16 v16h;
typedef __attribute__((ext_vector_type(8)))  _Float16 v8h;
typedef __attribute__((ext_vector_type(8)))  float    v8f;

// ---- CDNA5 async global->LDS path (guarded; sync fallback) ----------------
#if defined(__has_builtin)
#if __has_builtin(__builtin_amdgcn_global_load_async_to_lds_b128)
#define HAVE_ASYNC_LDS 1
#endif
#endif
#ifndef HAVE_ASYNC_LDS
#define HAVE_ASYNC_LDS 0
#endif

#if HAVE_ASYNC_LDS
// Builtin signature (from clang diagnostic): param0 = global int4*, then LDS
// int4*, then two integer immediates (offset, cpol).
typedef int v4i __attribute__((vector_size(16)));
typedef __attribute__((address_space(1))) v4i* gp_v4i;
typedef __attribute__((address_space(3))) v4i* lp_v4i;
static __device__ inline void cp16_g2l(const h16* g, h16* l) {
  // 16-byte async copy, tracked by ASYNCcnt (no VGPR round trip)
  __builtin_amdgcn_global_load_async_to_lds_b128((gp_v4i)g, (lp_v4i)l, 0, 0);
}
static __device__ inline void wait_async_all() {
#if __has_builtin(__builtin_amdgcn_s_wait_asynccnt)
  __builtin_amdgcn_s_wait_asynccnt(0);
#else
  asm volatile("s_wait_asynccnt 0x0" ::: "memory");
#endif
}
#else
static __device__ inline void cp16_g2l(const h16* g, h16* l) {
  *(v8h*)l = *(const v8h*)g;           // sync fallback: b128 load + ds store
}
static __device__ inline void wait_async_all() {}
#endif

static __device__ inline v8f wmma_f16(v16h a, v16h b, v8f c) {
  // D = A(16x32 f16) * B(32x16 f16) + C(16x16 f32)
  return __builtin_amdgcn_wmma_f32_16x16x32_f16(false, a, false, b,
                                                (short)0, c, false, false);
}

// Fragment load (A-pattern or B-from-row-major-W pattern).
// Caller passes p = base + row*ld + (lane>>4)*8 + ktile.
// halves 0..7  -> k = ktile + 8*kg + {0..7}
// halves 8..15 -> k = ktile + 16 + 8*kg + {0..7}
static __device__ inline v16h load_frag(const h16* p) {
  v8h lo = *(const v8h*)(p);
  v8h hi = *(const v8h*)(p + 16);
  v16h r;
#pragma unroll
  for (int i = 0; i < 8; ++i) { r[i] = lo[i]; r[i + 8] = hi[i]; }
  return r;
}

// ---------------------------------------------------------------------------
// 1) LayerNorm of hidden = x.transpose -> f16, LDS-tiled for coalescing.
//    x: (B, C=512, N=1024) f32; hn: (B*N, C) f16
//    Block 256 = 8 waves handles 16 n-rows; grid (N/16, B).
// ---------------------------------------------------------------------------
__global__ void k_layernorm(const float* __restrict__ x,
                            const float* __restrict__ w,
                            const float* __restrict__ bgain,
                            h16* __restrict__ hn) {
  __shared__ float tile[16][512 + 4];
  const int tid = threadIdx.x;
  const int lane = tid & 31;
  const int wave = tid >> 5;
  const int n0 = blockIdx.x * 16;
  const int b = blockIdx.y;
  const float* xb = x + (size_t)b * 512 * 1024 + n0;

  // coalesced load: consecutive tid -> consecutive n within a c-row of x
#pragma unroll 4
  for (int i = 0; i < 32; ++i) {
    int idx = i * 256 + tid;           // enumerates (c, n)
    int n = idx & 15, c = idx >> 4;
    tile[n][c] = xb[(size_t)c * 1024 + n];
  }
  __syncthreads();

  // one wave per 2 rows
#pragma unroll
  for (int j = 0; j < 2; ++j) {
    int n = wave * 2 + j;
    float vals[16];
    float sum = 0.f;
#pragma unroll
    for (int i = 0; i < 16; ++i) {
      vals[i] = tile[n][lane + 32 * i];
      sum += vals[i];
    }
#pragma unroll
    for (int off = 16; off; off >>= 1) sum += __shfl_xor(sum, off, 32);
    float mu = sum * (1.f / 512.f);
    float vs = 0.f;
#pragma unroll
    for (int i = 0; i < 16; ++i) { float d = vals[i] - mu; vs += d * d; }
#pragma unroll
    for (int off = 16; off; off >>= 1) vs += __shfl_xor(vs, off, 32);
    float rstd = rsqrtf(vs * (1.f / 512.f) + 1e-5f);

    h16* out = hn + ((size_t)(b * 1024 + n0 + n)) * 512;
#pragma unroll
    for (int i = 0; i < 16; ++i) {
      int c = lane + 32 * i;
      out[c] = (h16)((vals[i] - mu) * rstd * w[c] + bgain[c]);
    }
  }
}

// ---------------------------------------------------------------------------
// 2) f32 -> f16 cast (grid-stride)
// ---------------------------------------------------------------------------
__global__ void k_cast(const float* __restrict__ in, h16* __restrict__ out,
                       long n) {
  long i = (long)blockIdx.x * blockDim.x + threadIdx.x;
  long stride = (long)gridDim.x * blockDim.x;
  for (; i < n; i += stride) out[i] = (h16)in[i];
}

// ---------------------------------------------------------------------------
// 3) f16 WMMA GEMM with async-LDS-staged weight panel (double-buffered):
//      acc[m,n] = sum_k A[m,k] * W[n,k]
//    A: (M, K) f16 (ld=K); W: (Ncols, K) f16 (ld=K)
//    MODE 0: Out16[m, n] f16 row-major (ld = ldo)
//    MODE 1: Out32[b, c=n, nn=m%1024] = acc + X[same]  (fused residual +
//            transpose back to (B, C, H, W) f32)
//    Block 256 = 8 waves; wave tile 32Mx64N; grid (M/256, Ncols/64).
// ---------------------------------------------------------------------------
template <int MODE>
__global__ __launch_bounds__(256, 1)
void k_gemm(const h16* __restrict__ A, const h16* __restrict__ W,
            h16* __restrict__ Out16, int K, int ldo,
            const float* __restrict__ X, float* __restrict__ Out32) {
  const int tid = threadIdx.x;
  const int lane = tid & 31;
  const int wave = tid >> 5;
  const int kg = lane >> 4;
  const int l15 = lane & 15;
  const int m0 = blockIdx.x * 256 + wave * 32;
  const int n0 = blockIdx.y * 64;

  // double-buffered 64(row) x 32(k) f16 weight panel
  __shared__ __align__(16) h16 wbuf[2][64][32];

  // each of 256 threads stages one 16B chunk per panel
  const int srow = tid >> 2;           // 0..63
  const int sch = (tid & 3) * 8;       // half offset 0/8/16/24
  const h16* wsrc = W + (size_t)(n0 + srow) * K + sch;

  auto stage = [&](int s, int k) {
    cp16_g2l(wsrc + k, &wbuf[s][srow][sch]);
  };

  v8f acc[2][4] = {};
  const h16* ap0 = A + (size_t)(m0 + l15) * K + kg * 8;
  const h16* ap1 = ap0 + (size_t)16 * K;

  stage(0, 0);
  const int nsteps = K >> 5;
  for (int i = 0; i < nsteps; ++i) {
    const int k = i << 5;
    const int s = i & 1;
    wait_async_all();                  // my panel chunk landed in LDS
    __syncthreads();                   // everyone's chunk landed / prev reads done
    if (i + 1 < nsteps) stage(s ^ 1, k + 32);
    if (i + 4 < nsteps) {              // stream-prefetch A (global_prefetch_b8)
      __builtin_prefetch(ap0 + k + 128, 0, 1);
      __builtin_prefetch(ap1 + k + 128, 0, 1);
    }
    v16h a0 = load_frag(ap0 + k);
    v16h a1 = load_frag(ap1 + k);
#pragma unroll
    for (int j = 0; j < 4; ++j) {
      v16h bf = load_frag(&wbuf[s][j * 16 + l15][kg * 8]);  // ds_load_b128 x2
      acc[0][j] = wmma_f16(a0, bf, acc[0][j]);
      acc[1][j] = wmma_f16(a1, bf, acc[1][j]);
    }
  }

  // D layout: row = r + 8*kg (per 16-row sub-tile), col = l15 + 16*j
#pragma unroll
  for (int am = 0; am < 2; ++am) {
#pragma unroll
    for (int r = 0; r < 8; ++r) {
      int row = m0 + am * 16 + r + 8 * kg;
      if (MODE == 0) {
        size_t o = (size_t)row * ldo + n0;
        Out16[o +      l15] = (h16)acc[am][0][r];
        Out16[o + 16 + l15] = (h16)acc[am][1][r];
        Out16[o + 32 + l15] = (h16)acc[am][2][r];
        Out16[o + 48 + l15] = (h16)acc[am][3][r];
      } else {
        int bb = row >> 10;
        int nn = row & 1023;
        float vals[4] = {acc[am][0][r], acc[am][1][r], acc[am][2][r],
                         acc[am][3][r]};
#pragma unroll
        for (int j = 0; j < 4; ++j) {
          int c = n0 + j * 16 + l15;
          size_t oi = ((size_t)(bb * 512 + c)) * 1024 + nn;
          Out32[oi] = vals[j] + X[oi];   // residual = original x
        }
      }
    }
  }
}

// ---------------------------------------------------------------------------
// 4) V transpose via LDS tiles: V (B, L=512, C=512) -> VT (B, C, L), f16
//    Grid (L/64, C/64, B), block 256.
// ---------------------------------------------------------------------------
__global__ void k_transpose_v(const h16* __restrict__ V, h16* __restrict__ VT) {
  __shared__ h16 t[64][64 + 4];
  const int tid = threadIdx.x;
  const int b = blockIdx.z;
  const int e0 = blockIdx.y * 64;
  const int l0 = blockIdx.x * 64;
  const h16* vp = V + ((size_t)(b * 512 + l0)) * 512 + e0;
#pragma unroll 4
  for (int i = 0; i < 16; ++i) {
    int idx = i * 256 + tid;           // (l, e); e fastest -> coalesced read
    int e = idx & 63, l = idx >> 6;
    t[l][e] = vp[(size_t)l * 512 + e];
  }
  __syncthreads();
  h16* op = VT + ((size_t)(b * 512 + e0)) * 512 + l0;
#pragma unroll 4
  for (int i = 0; i < 16; ++i) {
    int idx = i * 256 + tid;           // (e, l); l fastest -> coalesced write
    int l = idx & 63, e = idx >> 6;
    op[(size_t)e * 512 + l] = t[l][e];
  }
}

// ---------------------------------------------------------------------------
// 5) Flash attention, one wave = 16 query rows of one (b, head).
//    Q: (B*N, C) f16; Km: (B*L, C) f16; VT: (B*C, L) f16; mask: (B, L) bool
//    O: (B*N, C) f16.  Grid: (N/128, NH, B), block = 256.
//    launch_bounds(256,1): full VGPR budget -> no scratch spill of Q frags.
// ---------------------------------------------------------------------------
__global__ __launch_bounds__(256, 1)
void k_attn(const h16* __restrict__ Q, const h16* __restrict__ Km,
            const h16* __restrict__ VT,
            const unsigned char* __restrict__ mask,
            h16* __restrict__ O) {
  const int lane = threadIdx.x & 31;
  const int wave = threadIdx.x >> 5;
  const int kg = lane >> 4;
  const int l15 = lane & 15;
  const int b = blockIdx.z;
  const int h = blockIdx.y;
  const int m0 = blockIdx.x * 128 + wave * 16;

  __shared__ __align__(16) h16 pl[8][16][32];   // per-wave f16 P tile (16x32)

  const h16* qp = Q + ((size_t)(b * 1024 + m0 + l15)) * 512 + h * 64 + kg * 8;
  v16h qa0 = load_frag(qp);
  v16h qa1 = load_frag(qp + 32);

  float mrow[8], lrow[8];
  v8f o0 = {}, o1 = {}, o2 = {}, o3 = {};
#pragma unroll
  for (int r = 0; r < 8; ++r) { mrow[r] = -1e30f; lrow[r] = 0.f; }

  const h16* kbase = Km + ((size_t)(b * 512 + l15)) * 512 + h * 64 + kg * 8;
  const h16* vbase = VT + ((size_t)(b * 512 + h * 64 + l15)) * 512 + kg * 8;
  const unsigned char* mk = mask + (size_t)b * 512;

  for (int ls = 0; ls < 512; ls += 32) {
    v8f sst[2];
#pragma unroll
    for (int lt = 0; lt < 2; ++lt) {
      const h16* kp = kbase + (size_t)(ls + lt * 16) * 512;
      v16h kb0 = load_frag(kp);
      v16h kb1 = load_frag(kp + 32);
      v8f s = {};
      s = wmma_f16(qa0, kb0, s);
      s = wmma_f16(qa1, kb1, s);
      int lcol = ls + lt * 16 + l15;
      bool ok = mk[lcol] != 0;
#pragma unroll
      for (int r = 0; r < 8; ++r) s[r] = ok ? s[r] * 0.125f : -1e30f;
      sst[lt] = s;
    }
    // online softmax; stats align with C-layout rows (r + 8*kg)
#pragma unroll
    for (int r = 0; r < 8; ++r) {
      float mx = fmaxf(sst[0][r], sst[1][r]);
#pragma unroll
      for (int off = 8; off; off >>= 1) mx = fmaxf(mx, __shfl_xor(mx, off, 16));
      float mn = fmaxf(mrow[r], mx);
      float corr = __expf(mrow[r] - mn);
      float p0 = __expf(sst[0][r] - mn);
      float p1 = __expf(sst[1][r] - mn);
      float ps = p0 + p1;
#pragma unroll
      for (int off = 8; off; off >>= 1) ps += __shfl_xor(ps, off, 16);
      lrow[r] = lrow[r] * corr + ps;
      mrow[r] = mn;
      o0[r] *= corr; o1[r] *= corr; o2[r] *= corr; o3[r] *= corr;
      pl[wave][r + 8 * kg][l15]      = (h16)p0;   // DS in-order within wave
      pl[wave][r + 8 * kg][16 + l15] = (h16)p1;
    }
    // P (16x32) x V (32x64): reload P in A-fragment layout from LDS
    v16h pa = load_frag(&pl[wave][l15][kg * 8]);
    const h16* vp = vbase + ls;
    v16h v0 = load_frag(vp);
    v16h v1 = load_frag(vp + (size_t)16 * 512);
    v16h v2 = load_frag(vp + (size_t)32 * 512);
    v16h v3 = load_frag(vp + (size_t)48 * 512);
    o0 = wmma_f16(pa, v0, o0);
    o1 = wmma_f16(pa, v1, o1);
    o2 = wmma_f16(pa, v2, o2);
    o3 = wmma_f16(pa, v3, o3);
  }
#pragma unroll
  for (int r = 0; r < 8; ++r) {
    float inv = 1.f / lrow[r];
    size_t row = ((size_t)(b * 1024 + m0 + r + 8 * kg)) * 512 + h * 64;
    O[row +      l15] = (h16)(o0[r] * inv);
    O[row + 16 + l15] = (h16)(o1[r] * inv);
    O[row + 32 + l15] = (h16)(o2[r] * inv);
    O[row + 48 + l15] = (h16)(o3[r] * inv);
  }
}

// ---------------------------------------------------------------------------
// Host launcher
// ---------------------------------------------------------------------------
extern "C" void kernel_launch(void* const* d_in, const int* in_sizes, int n_in,
                              void* d_out, int out_size, void* d_ws, size_t ws_size,
                              hipStream_t stream) {
  (void)in_sizes; (void)n_in; (void)out_size; (void)ws_size;
  const float* x      = (const float*)d_in[0];
  const float* ctx32  = (const float*)d_in[1];
  const unsigned char* cmask = (const unsigned char*)d_in[2]; // bool mask
  const float* lnw    = (const float*)d_in[3];
  const float* lnb    = (const float*)d_in[4];
  const float* Wq32   = (const float*)d_in[5];
  const float* Wk32   = (const float*)d_in[6];
  const float* Wv32   = (const float*)d_in[7];
  const float* Wo32   = (const float*)d_in[8];
  const float* Wctx32 = (const float*)d_in[9];

  char* ws = (char*)d_ws;
  size_t off = 0;
  auto carve = [&](size_t bytes) -> void* {
    void* p = ws + off;
    off = (off + bytes + 255) & ~(size_t)255;
    return p;
  };
  h16* hn     = (h16*)carve(16384ull * 512 * 2);  // hidden_norm f16
  h16* cxt16  = (h16*)carve(8192ull * 768 * 2);   // context f16
  h16* wq16   = (h16*)carve(512ull * 512 * 2);
  h16* wk16   = (h16*)carve(512ull * 512 * 2);
  h16* wv16   = (h16*)carve(512ull * 512 * 2);
  h16* wo16   = (h16*)carve(512ull * 512 * 2);
  h16* wctx16 = (h16*)carve(512ull * 768 * 2);
  h16* cp16   = (h16*)carve(8192ull * 512 * 2);   // ctx projected
  h16* q16    = (h16*)carve(16384ull * 512 * 2);
  h16* k16    = (h16*)carve(8192ull * 512 * 2);
  h16* v16p   = (h16*)carve(8192ull * 512 * 2);
  h16* vt16   = (h16*)carve(8192ull * 512 * 2);   // V transposed (B, C, L)
  h16* ao16   = (h16*)carve(16384ull * 512 * 2);  // attention output

  // 1) LayerNorm -> hidden_norm f16
  k_layernorm<<<dim3(64, 16), 256, 0, stream>>>(x, lnw, lnb, hn);
  // 2) casts
  k_cast<<<1024, 256, 0, stream>>>(ctx32,  cxt16,  8192l * 768);
  k_cast<<<256,  256, 0, stream>>>(Wq32,   wq16,   512l * 512);
  k_cast<<<256,  256, 0, stream>>>(Wk32,   wk16,   512l * 512);
  k_cast<<<256,  256, 0, stream>>>(Wv32,   wv16,   512l * 512);
  k_cast<<<256,  256, 0, stream>>>(Wo32,   wo16,   512l * 512);
  k_cast<<<384,  256, 0, stream>>>(Wctx32, wctx16, 512l * 768);
  // 3) GEMMs (WMMA f16, async-LDS-staged weight panels)
  k_gemm<0><<<dim3(32, 8), 256, 0, stream>>>(cxt16, wctx16, cp16, 768, 512,
                                             nullptr, nullptr);          // ctx
  k_gemm<0><<<dim3(64, 8), 256, 0, stream>>>(hn, wq16, q16, 512, 512,
                                             nullptr, nullptr);          // Q
  k_gemm<0><<<dim3(32, 8), 256, 0, stream>>>(cp16, wk16, k16, 512, 512,
                                             nullptr, nullptr);          // K
  k_gemm<0><<<dim3(32, 8), 256, 0, stream>>>(cp16, wv16, v16p, 512, 512,
                                             nullptr, nullptr);          // V
  // 4) V transpose for contiguous B-fragments in attention
  k_transpose_v<<<dim3(8, 8, 16), 256, 0, stream>>>(v16p, vt16);
  // 5) flash attention
  k_attn<<<dim3(8, 8, 16), 256, 0, stream>>>(q16, k16, vt16, cmask, ao16);
  // 6) out projection + residual + transpose to (B, C, H, W)
  k_gemm<1><<<dim3(64, 8), 256, 0, stream>>>(ao16, wo16, nullptr, 512, 0,
                                             x, (float*)d_out);
}
